// SpodNet_7095285973283
// MI455X (gfx1250) — matrix operand order repaired and checked by reading, not compiled, not run on hardware
//
#include <hip/hip_runtime.h>

// ---------------------------------------------------------------------------
// SpodNet on MI455X (gfx1250), dead-code-eliminated:
//   out[b,i,j] = Theta0[i,j] + 129*(t12n[b,max(i,j),min(i,j)] - Theta0[min,max])  (i!=j)
//   out[b,i,i] = Theta0[i,i]
// t12n[b,col,:] = ReLU(feats(b,col) @ W1 + b1) @ W2 + b2, feats gathered from
// Theta0[b] with row/col `col` removed.  Heavy op: [131072 x 16257] virtual
// GEMM -> v_wmma_f32_16x16x32_bf16, fp32 accum.  B-fragments are register
// double-buffered so WMMAs never wait on same-iteration global loads.
// ---------------------------------------------------------------------------

#define PDIM 128
#define HDIM 127
#define KSTEPS_T11 508    // 508*32 = 16256 = 127*128 (T11 padded)
#define TH_STRIDE 132     // u16 elems; 264B rows -> bank-shifted Theta gather
#define Z_STRIDE  136     // u16 elems; 272B rows -> 16B-aligned ds_load_b128
#define KP1 16416         // padded K for GEMM1 = 513 * 32

typedef __attribute__((ext_vector_type(16))) __bf16 v16bf;
typedef __attribute__((ext_vector_type(8)))  float  v8f;

union Frag16 { uint4 q[2]; unsigned int u[8]; unsigned short h[16]; v16bf v; };
union Acc8   { v8f v; float f[8]; };

__device__ __forceinline__ unsigned short f2bf(float f) {
    unsigned x = __builtin_bit_cast(unsigned, f);
    x += 0x7FFFu + ((x >> 16) & 1u);            // round-to-nearest-even
    return (unsigned short)(x >> 16);
}

// element e (0..15) of a 16-bit A fragment -> K offset (plus 8*g)
#define AKOFF(e) (((e) & 7) + (((e) >> 3) << 4))

__device__ __forceinline__ void loadB8(Frag16* bfr, const unsigned short* base) {
    const int l = threadIdx.x & 31;
    const unsigned short* bb = base - (size_t)l * PDIM;
    const int hh = l >> 4, nn = l & 15;
#pragma unroll
    for (int nt = 0; nt < 8; ++nt) {
#pragma unroll
        for (int s = 0; s < 16; ++s)
            bfr[nt].h[s] = bb[(size_t)(AKOFF(s) + (hh << 3)) * PDIM + nt * 16 + nn];
    }
}

__device__ __forceinline__ void wmma8(Acc8* acc, const Frag16& af, const Frag16* bfr) {
#pragma unroll
    for (int nt = 0; nt < 8; ++nt)
        acc[nt].v = __builtin_amdgcn_wmma_f32_16x16x32_bf16(
            false, af.v, false, bfr[nt].v, (short)0, acc[nt].v, false, false);
}

// branch-free T11 gather: one (uniform) row per k-step, idxm shifts map the
// padded element (col 127+shift -> 128) onto the zeroed LDS pad column.
__device__ __forceinline__ Frag16 gatherT11(const unsigned short* sT, int kt,
                                            int colv, int g) {
    const int r     = kt >> 2;
    const int ri    = r + (r >= colv);
    const int rowA  = ri * TH_STRIDE;
    const int kbase = ((kt & 3) << 5) + (g << 3);
    Frag16 af;
#pragma unroll
    for (int e = 0; e < 16; ++e) {
        const int cc = kbase + AKOFF(e);
        const int c  = cc + (cc >= colv);
        af.h[e] = sT[rowA + c];
    }
    return af;
}

// ---- prep: W1 [16257 x 127] f32 -> W1p [16416 x 128] bf16, permuted K ------
__global__ void spod_prep_w1(const float* __restrict__ W1,
                             unsigned short* __restrict__ W1p) {
    int idx = blockIdx.x * blockDim.x + threadIdx.x;      // KP1*128 threads
    if (idx >= KP1 * PDIM) return;
    int kp = idx >> 7;
    int n  = idx & 127;
    float val = 0.f;
    if (n < HDIM) {
        int ok = -1;
        if (kp < 16384) {
            int r = kp >> 7, cc = kp & 127;
            if (cc < HDIM) ok = (r < HDIM) ? (r * HDIM + cc) : (16129 + cc);
        } else if (kp == 16384) {
            ok = 16256;                                   // t22 feature
        }
        if (ok >= 0) val = W1[ok * HDIM + n];
    }
    W1p[idx] = f2bf(val);
}

__global__ void spod_prep_w2(const float* __restrict__ W2,
                             unsigned short* __restrict__ W2p) {
    int idx = blockIdx.x * blockDim.x + threadIdx.x;      // 128*128 threads
    if (idx >= PDIM * PDIM) return;
    int k = idx >> 7, n = idx & 127;
    float val = (k < HDIM && n < HDIM) ? W2[k * HDIM + n] : 0.f;
    W2p[idx] = f2bf(val);
}

__global__ void spod_prep_bias(const float* __restrict__ b1,
                               const float* __restrict__ b2,
                               float* __restrict__ b1p,
                               float* __restrict__ b2p) {
    int n = threadIdx.x;
    if (n < PDIM) {
        b1p[n] = (n < HDIM) ? b1[n] : 0.f;
        b2p[n] = (n < HDIM) ? b2[n] : 0.f;
    }
}

// ---- main: one workgroup (8 wave32) per batch ------------------------------
__global__ __launch_bounds__(256)
void spod_main(const float* __restrict__ Theta,
               const unsigned short* __restrict__ W1p,
               const unsigned short* __restrict__ W2p,
               const float* __restrict__ b1p,
               const float* __restrict__ b2p,
               float* __restrict__ out) {
    __shared__ unsigned short sT[17408];   // Theta(129x132) then Z(128x136)

    const int b    = blockIdx.x;
    const int tid  = threadIdx.x;
    const int lane = tid & 31;
    const int wave = tid >> 5;
    const int g    = lane >> 4;            // half-wave id
    const int ln   = lane & 15;
    const float* Tg = Theta + (size_t)b * PDIM * PDIM;
    float*       Og = out   + (size_t)b * PDIM * PDIM;

    // zero pad row 128 (t12 pad target) and pad column 128 (T11 pad target)
    if (tid < 132 + 129) {
        if (tid < 132) sT[128 * TH_STRIDE + tid] = 0;
        else           sT[(tid - 132) * TH_STRIDE + 128] = 0;
    }
    // stage Theta0[b] as bf16 into LDS
    for (int i = tid; i < PDIM * PDIM; i += 256) {
        int r = i >> 7, c = i & 127;
        sT[r * TH_STRIDE + c] = f2bf(Tg[i]);
    }

    const int rowBase = wave * 16;         // this wave's 16 sweep-columns (M dim)
    const int colv    = rowBase + ln;      // per-lane A-row = removed column

    Acc8 acc[8];
#pragma unroll
    for (int nt = 0; nt < 8; ++nt) acc[nt].v = (v8f){0,0,0,0,0,0,0,0};

    // prologue B load overlaps the staging barrier
    Frag16 b0[8], b1[8];
    loadB8(b0, W1p + (size_t)lane * PDIM);
    __syncthreads();

    // ---------------- GEMM1 phase A: T11, double-buffered -------------------
    for (int kt = 0; kt < KSTEPS_T11; kt += 2) {
        const unsigned short* Bn1 = W1p + (size_t)((kt + 1) * 32 + lane) * PDIM;
        __builtin_prefetch(Bn1 + 96 * PDIM, 0, 3);        // ~3 slabs ahead
        loadB8(b1, Bn1);                                  // next iter's B in flight
        Frag16 af = gatherT11(sT, kt, colv, g);
        wmma8(acc, af, b0);                               // consumes prior loads

        loadB8(b0, W1p + (size_t)((kt + 2) * 32 + lane) * PDIM); // rows <= 16287, in-bounds
        af = gatherT11(sT, kt + 1, colv, g);
        wmma8(acc, af, b1);
    }

    // ---------------- GEMM1 phase B: t12 rows (kt = 508..511) ---------------
#pragma unroll
    for (int kt = KSTEPS_T11; kt < 512; ++kt) {
        const int kbase = ((kt & 3) << 5) + (g << 3);
        Frag16 af;
#pragma unroll
        for (int e = 0; e < 16; ++e) {
            const int cc = kbase + AKOFF(e);              // t12 element index
            const int rr = cc + (cc >= colv);             // idxm row (127->128 pad row)
            af.h[e] = sT[rr * TH_STRIDE + colv];
        }
        Frag16 bfr[8];
        loadB8(bfr, W1p + (size_t)(kt * 32 + lane) * PDIM);
        wmma8(acc, af, bfr);
    }

    // ---------------- GEMM1 phase C: t22 singleton (kt = 512) ---------------
    {
        Frag16 af;
#pragma unroll
        for (int e = 0; e < 16; ++e) af.h[e] = 0;
        if (g == 0) af.h[0] = sT[colv * TH_STRIDE + colv];   // K=16384 -> t22
        Frag16 bfr[8];
        loadB8(bfr, W1p + (size_t)(16384 + lane) * PDIM);
        wmma8(acc, af, bfr);
    }

    // bias + ReLU -> Z (bf16) into LDS, reusing the Theta region
    __syncthreads();                        // everyone done reading sT as Theta
#pragma unroll
    for (int nt = 0; nt < 8; ++nt) {
        const int n  = nt * 16 + ln;
        const float bb = b1p[n];
#pragma unroll
        for (int v = 0; v < 8; ++v) {
            const int m = rowBase + v + (g << 3);          // C layout: M = v + 8*g
            float z = acc[nt].f[v] + bb;
            z = z > 0.f ? z : 0.f;
            sT[m * Z_STRIDE + n] = f2bf(z);
        }
    }
    __syncthreads();

    // ---------------- GEMM2: Z @ W2p, K = 128 -------------------------------
    Acc8 acc2[8];
#pragma unroll
    for (int nt = 0; nt < 8; ++nt) acc2[nt].v = (v8f){0,0,0,0,0,0,0,0};

#pragma unroll
    for (int kt = 0; kt < 4; ++kt) {
        const int k0 = kt * 32;
        Frag16 af2;
        const int arow = rowBase + ln;
        af2.q[0] = *(const uint4*)&sT[arow * Z_STRIDE + k0 + (g << 3)];
        af2.q[1] = *(const uint4*)&sT[arow * Z_STRIDE + k0 + 16 + (g << 3)];
        Frag16 bfr[8];
        loadB8(bfr, W2p + (size_t)(k0 + lane) * PDIM);
        wmma8(acc2, af2, bfr);
    }

    // ---------------- epilogue: out = Theta0 + 129 * Delta ------------------
#pragma unroll
    for (int nt = 0; nt < 8; ++nt) {
        const int k   = nt * 16 + ln;                      // min index / t12n slot
        const float bb2 = b2p[k];
#pragma unroll
        for (int v = 0; v < 8; ++v) {
            const int m = rowBase + v + (g << 3);          // max index (sweep col)
            const float t = acc2[nt].f[v] + bb2;           // t12n[b, m, k]
            if (k < m) {
                const float tmk = Tg[m * PDIM + k];
                const float tkm = Tg[k * PDIM + m];
                const float d   = t - tkm;
                Og[m * PDIM + k] = tmk + 129.f * d;
                Og[k * PDIM + m] = tkm + 129.f * d;
            } else if (k == m) {
                Og[m * PDIM + m] = Tg[m * PDIM + m];       // diagonal untouched
            }
        }
    }
}

// ---------------------------------------------------------------------------
extern "C" void kernel_launch(void* const* d_in, const int* in_sizes, int n_in,
                              void* d_out, int out_size, void* d_ws, size_t ws_size,
                              hipStream_t stream) {
    const float* Theta = (const float*)d_in[0];
    const float* W1    = (const float*)d_in[1];
    const float* b1    = (const float*)d_in[2];
    const float* W2    = (const float*)d_in[3];
    const float* b2    = (const float*)d_in[4];
    float* out = (float*)d_out;

    char* ws = (char*)d_ws;
    unsigned short* W1p = (unsigned short*)ws;                       // 16416*128*2 = 4202496 B
    unsigned short* W2p = (unsigned short*)(ws + 4202496);           // 32768 B
    float*          b1p = (float*)(ws + 4202496 + 32768);            // 512 B
    float*          b2p = (float*)(ws + 4202496 + 32768 + 512);      // 512 B

    spod_prep_w1<<<(KP1 * PDIM) / 256, 256, 0, stream>>>(W1, W1p);
    spod_prep_w2<<<(PDIM * PDIM) / 256, 256, 0, stream>>>(W2, W2p);
    spod_prep_bias<<<1, 256, 0, stream>>>(b1, b2, b1p, b2p);
    spod_main<<<1024, 256, 0, stream>>>(Theta, W1p, W2p, b1p, b2p, out);
}
